// Attention_10411000725842
// MI455X (gfx1250) — compile-verified, matrix operand order checked by reading, not run on hardware
//
#include <hip/hip_runtime.h>
#include <hip/hip_bf16.h>
#include <math.h>

// ---------------------------------------------------------------------------
// Types for CDNA5 WMMA (wave32): D(16x16 f32) = A(16x32 f16) x B(32x16 f16) + C
// ---------------------------------------------------------------------------
typedef __attribute__((ext_vector_type(16))) _Float16 v16h;
typedef __attribute__((ext_vector_type(8)))  _Float16 v8h;   // 16-byte load unit
typedef __attribute__((ext_vector_type(8)))  float    v8f;
typedef int b128_t __attribute__((vector_size(16)));         // matches async-LDS builtin pointee

#define WMMA_F16(a, b, c) \
  __builtin_amdgcn_wmma_f32_16x16x32_f16(false, (a), false, (b), (short)0, (c), false, false)

// Problem constants
#define BATCH   4
#define SEQ     2048
#define DIM     1024
#define HEADS   16
#define DHEAD   64
#define TOKENS  (BATCH * SEQ)      // 8192
#define QKVCOLS (3 * DIM)          // 3072

// Async global->LDS path (gfx1250). Guarded so either toolchain compiles.
#if defined(__AMDGCN__) && __has_builtin(__builtin_amdgcn_global_load_async_to_lds_b128) && \
    __has_builtin(__builtin_amdgcn_s_wait_asynccnt)
#define USE_ASYNC_LDS 1
#else
#define USE_ASYNC_LDS 0
#endif

// Copy 16 bytes global -> LDS (async when available).
__device__ inline void copy16_g2l(const _Float16* __restrict__ g, _Float16* __restrict__ l) {
#if USE_ASYNC_LDS
    __builtin_amdgcn_global_load_async_to_lds_b128(
        (__attribute__((address_space(1))) b128_t*)g,
        (__attribute__((address_space(3))) b128_t*)l, 0, 0);
#else
    *(v8h*)l = *(const v8h*)g;
#endif
}

__device__ inline void wait_copies() {
#if USE_ASYNC_LDS
    __builtin_amdgcn_s_wait_asynccnt(0);
#endif
}

// Load one 16x32 f16 fragment (A layout; also B layout when source is row-major N x K,
// i.e. the transposed operand). Lane l: row = l&15, kbase = 8*(l>>4);
// lo covers K[kbase .. kbase+7], hi covers K[16+kbase .. 16+kbase+7].
__device__ inline v16h load_frag16(const _Float16* __restrict__ base, int stride, int lane) {
    const int r  = lane & 15;
    const int kb = (lane >> 4) << 3;
    const _Float16* p = base + (size_t)r * stride + kb;
    v8h lo = *(const v8h*)(p);
    v8h hi = *(const v8h*)(p + 16);
    v16h out;
#pragma unroll
    for (int e = 0; e < 8; ++e) { out[e] = lo[e]; out[e + 8] = hi[e]; }
    return out;
}

// ---------------------------------------------------------------------------
// 1) Weight transpose + f32 -> f16 cast:  W[K][N] -> WT[N][K]
// ---------------------------------------------------------------------------
__global__ __launch_bounds__(256) void k_transpose_cast(const float* __restrict__ W,
                                                        _Float16* __restrict__ WT,
                                                        int K, int N) {
    int idx = blockIdx.x * 256 + threadIdx.x;
    if (idx >= K * N) return;
    int n = idx / K;
    int k = idx - n * K;
    WT[idx] = (_Float16)W[(size_t)k * N + n];
}

// ---------------------------------------------------------------------------
// 2) LayerNorm (fp32 math) -> f16 activations. One block per token row.
// ---------------------------------------------------------------------------
__global__ __launch_bounds__(256) void k_layernorm(const float* __restrict__ x,
                                                   const float* __restrict__ g,
                                                   const float* __restrict__ bta,
                                                   _Float16* __restrict__ xn) {
    __shared__ float s1[256], s2[256];
    const int row = blockIdx.x;
    const float* xr = x + (size_t)row * DIM;
    float vals[4];
    float a = 0.f, sq = 0.f;
#pragma unroll
    for (int t = 0; t < 4; ++t) {
        float v = xr[threadIdx.x + 256 * t];
        vals[t] = v; a += v; sq += v * v;
    }
    s1[threadIdx.x] = a; s2[threadIdx.x] = sq;
    __syncthreads();
    for (int off = 128; off > 0; off >>= 1) {
        if (threadIdx.x < off) {
            s1[threadIdx.x] += s1[threadIdx.x + off];
            s2[threadIdx.x] += s2[threadIdx.x + off];
        }
        __syncthreads();
    }
    const float mean = s1[0] * (1.f / DIM);
    const float var  = s2[0] * (1.f / DIM) - mean * mean;
    const float rstd = rsqrtf(var + 1e-5f);
    _Float16* xo = xn + (size_t)row * DIM;
#pragma unroll
    for (int t = 0; t < 4; ++t) {
        int i = threadIdx.x + 256 * t;
        xo[i] = (_Float16)((vals[t] - mean) * rstd * g[i] + bta[i]);
    }
}

// ---------------------------------------------------------------------------
// 3) QKV GEMM: [8192 x 1024] x [1024 x 3072] -> Q [B,H,N,D], K [B,H,N,D], Vt [B,H,D,N]
//    One wave per 64x64 output tile; WT is pre-transposed so A and B loads match.
// ---------------------------------------------------------------------------
__global__ __launch_bounds__(32) void k_qkv_gemm(const _Float16* __restrict__ Xn,
                                                 const _Float16* __restrict__ WT,
                                                 _Float16* __restrict__ Q,
                                                 _Float16* __restrict__ Kd,
                                                 _Float16* __restrict__ Vt) {
    const int m0 = blockIdx.x * 64;
    const int n0 = blockIdx.y * 64;
    const int lane = threadIdx.x;

    v8f acc[4][4] = {};
    for (int k0 = 0; k0 < DIM; k0 += 32) {
        v16h a[4], b[4];
#pragma unroll
        for (int i = 0; i < 4; ++i)
            a[i] = load_frag16(Xn + (size_t)(m0 + 16 * i) * DIM + k0, DIM, lane);
#pragma unroll
        for (int j = 0; j < 4; ++j)
            b[j] = load_frag16(WT + (size_t)(n0 + 16 * j) * DIM + k0, DIM, lane);
        if (k0 + 32 < DIM) {
            __builtin_prefetch(Xn + (size_t)(m0 + (lane & 15)) * DIM + k0 + 32, 0, 3);
            __builtin_prefetch(WT + (size_t)(n0 + (lane & 15)) * DIM + k0 + 32, 0, 3);
        }
#pragma unroll
        for (int i = 0; i < 4; ++i)
#pragma unroll
            for (int j = 0; j < 4; ++j)
                acc[i][j] = WMMA_F16(a[i], b[j], acc[i][j]);
    }

    // Epilogue: scatter into Q / K (row-major per head) and V transposed.
    const int h2 = (lane >> 4) << 3;    // 0 or 8
    const int cl = lane & 15;
#pragma unroll
    for (int i = 0; i < 4; ++i)
#pragma unroll
        for (int j = 0; j < 4; ++j)
#pragma unroll
            for (int v = 0; v < 8; ++v) {
                int r = m0 + 16 * i + h2 + v;          // global token
                int c = n0 + 16 * j + cl;              // qkv column
                float val = acc[i][j][v];
                int which = c >> 10;
                int head  = (c >> 6) & 15;
                int d     = c & 63;
                int bb    = r >> 11;
                int n     = r & 2047;
                int bh    = bb * HEADS + head;
                if (which == 0)
                    Q[((size_t)bh * SEQ + n) * DHEAD + d] = (_Float16)val;
                else if (which == 1)
                    Kd[((size_t)bh * SEQ + n) * DHEAD + d] = (_Float16)val;
                else
                    Vt[((size_t)bh * DHEAD + d) * SEQ + n] = (_Float16)val;
            }
}

// ---------------------------------------------------------------------------
// 4) Flash attention, LDS-cooperative version.
//    4 waves / block, one (b,h) per block, 128 queries per block (32 per wave).
//    K/V tiles (32 keys) are double-buffered in LDS and shared by all 4 waves,
//    cutting global K/V traffic 8x vs one-wave-per-16-queries.
//    S^T = K x Q^T puts one query row per lane -> exp() packs straight into the
//    A-fragment of the P x V WMMA.
// ---------------------------------------------------------------------------
#define KT_PAD 72   // 64 + 8 f16 -> 144B row stride (bank spread)
#define VT_PAD 40   // 32 + 8 f16 ->  80B row stride
#define KEYS_PER_TILE 32

__global__ __launch_bounds__(128) void k_attention(const _Float16* __restrict__ Q,
                                                   const _Float16* __restrict__ Kd,
                                                   const _Float16* __restrict__ Vt,
                                                   _Float16* __restrict__ AO) {
    __shared__ __align__(16) _Float16 sK[2][KEYS_PER_TILE * KT_PAD]; // 2 x 4608 B
    __shared__ __align__(16) _Float16 sV[2][DHEAD * VT_PAD];         // 2 x 5120 B

    const int tid  = threadIdx.x;
    const int wave = tid >> 5;
    const int lane = tid & 31;
    const int blk  = blockIdx.x;           // 1024 = (B*H) * (SEQ/128)
    const int qblk = blk & 15;
    const int bh   = blk >> 4;
    const int bb   = bh >> 4;
    const int hd   = bh & 15;
    const int q0   = qblk * 128 + wave * 32;   // this wave's first query
    const int h2   = (lane >> 4) << 3;         // 0 or 8
    const int cl   = lane & 15;

    const _Float16* Qp = Q  + ((size_t)bh * SEQ + q0) * DHEAD;
    const _Float16* Kp = Kd + (size_t)bh * SEQ * DHEAD;
    const _Float16* Vp = Vt + (size_t)bh * DHEAD * SEQ;

    // Cooperative tile copy: 128 threads x 2 x 16B chunks for each of K and V.
    // K tile: 32 rows x 64 f16 (8 chunks/row). V tile: 64 rows x 32 f16 (4 chunks/row).
    const int ck0 = tid, ck1 = tid + 128;      // 256 chunks each

    // Preload Q fragments (B operand of S^T gemm), fold in softmax scale 1/8.
    v16h qf[2][2];
#pragma unroll
    for (int qb = 0; qb < 2; ++qb) {
        qf[qb][0] = load_frag16(Qp + (size_t)(16 * qb) * DHEAD,      DHEAD, lane);
        qf[qb][1] = load_frag16(Qp + (size_t)(16 * qb) * DHEAD + 32, DHEAD, lane);
#pragma unroll
        for (int e = 0; e < 16; ++e) {
            qf[qb][0][e] = (_Float16)((float)qf[qb][0][e] * 0.125f);
            qf[qb][1][e] = (_Float16)((float)qf[qb][1][e] * 0.125f);
        }
    }

    // Prime buffer 0 with key tile 0.
    {
        int r, o;
        r = ck0 >> 3; o = (ck0 & 7) * 8;
        copy16_g2l(Kp + (size_t)r * DHEAD + o, &sK[0][r * KT_PAD + o]);
        r = ck1 >> 3; o = (ck1 & 7) * 8;
        copy16_g2l(Kp + (size_t)r * DHEAD + o, &sK[0][r * KT_PAD + o]);
        r = ck0 >> 2; o = (ck0 & 3) * 8;
        copy16_g2l(Vp + (size_t)r * SEQ + o, &sV[0][r * VT_PAD + o]);
        r = ck1 >> 2; o = (ck1 & 3) * 8;
        copy16_g2l(Vp + (size_t)r * SEQ + o, &sV[0][r * VT_PAD + o]);
    }
    wait_copies();
    __syncthreads();

    v8f o[2][4] = {};                          // [qblock][dtile], 16 rows x 64 d each
    float m_run[2] = {-1e30f, -1e30f};
    float l_run[2] = {0.f, 0.f};

    const int NITER = SEQ / KEYS_PER_TILE;     // 64
    for (int it = 0; it < NITER; ++it) {
        const int cur = it & 1;

        // Issue next tile's copies into the other buffer (overlaps with compute).
        if (it + 1 < NITER) {
            const int j1 = (it + 1) * KEYS_PER_TILE;
            _Float16* dK = sK[cur ^ 1];
            _Float16* dV = sV[cur ^ 1];
            int r, oo;
            r = ck0 >> 3; oo = (ck0 & 7) * 8;
            copy16_g2l(Kp + (size_t)(j1 + r) * DHEAD + oo, &dK[r * KT_PAD + oo]);
            r = ck1 >> 3; oo = (ck1 & 7) * 8;
            copy16_g2l(Kp + (size_t)(j1 + r) * DHEAD + oo, &dK[r * KT_PAD + oo]);
            r = ck0 >> 2; oo = (ck0 & 3) * 8;
            copy16_g2l(Vp + (size_t)r * SEQ + j1 + oo, &dV[r * VT_PAD + oo]);
            r = ck1 >> 2; oo = (ck1 & 3) * 8;
            copy16_g2l(Vp + (size_t)r * SEQ + j1 + oo, &dV[r * VT_PAD + oo]);
        }

        // ---- S^T tiles from LDS: A = K rows, B = Q^T (shared across both q-blocks)
        const _Float16* bk = sK[cur];
        v16h kf0a = load_frag16(bk,                    KT_PAD, lane);
        v16h kf0b = load_frag16(bk + 32,               KT_PAD, lane);
        v16h kf1a = load_frag16(bk + 16 * KT_PAD,      KT_PAD, lane);
        v16h kf1b = load_frag16(bk + 16 * KT_PAD + 32, KT_PAD, lane);

        v16h pf[2];
#pragma unroll
        for (int qb = 0; qb < 2; ++qb) {
            v8f s0 = {}; s0 = WMMA_F16(kf0a, qf[qb][0], s0); s0 = WMMA_F16(kf0b, qf[qb][1], s0);
            v8f s1 = {}; s1 = WMMA_F16(kf1a, qf[qb][0], s1); s1 = WMMA_F16(kf1b, qf[qb][1], s1);
            // lane: query row cl of this q-block; keys j0 + v + 8*(lane>>4) (s0), +16 (s1)

            float mloc = -1e30f;
#pragma unroll
            for (int v = 0; v < 8; ++v) { mloc = fmaxf(mloc, s0[v]); mloc = fmaxf(mloc, s1[v]); }
            mloc = fmaxf(mloc, __shfl_xor(mloc, 16, 32));
            const float m_new = fmaxf(m_run[qb], mloc);

            float e0[8], e1[8];
            float rs = 0.f;
#pragma unroll
            for (int v = 0; v < 8; ++v) {
                e0[v] = __expf(s0[v] - m_new);
                e1[v] = __expf(s1[v] - m_new);
                rs += e0[v] + e1[v];
            }
            rs += __shfl_xor(rs, 16, 32);
            const float alpha = __expf(m_run[qb] - m_new);
            l_run[qb] = l_run[qb] * alpha + rs;
            m_run[qb] = m_new;

            // Rescale O accumulators (O lane holds rows h2+v -> gather row alphas).
            float av[8];
#pragma unroll
            for (int v = 0; v < 8; ++v) av[v] = __shfl(alpha, h2 + v, 32);
#pragma unroll
            for (int dt = 0; dt < 4; ++dt)
#pragma unroll
                for (int v = 0; v < 8; ++v) o[qb][dt][v] *= av[v];

            // P is already in A-fragment orientation.
#pragma unroll
            for (int e = 0; e < 8; ++e) { pf[qb][e] = (_Float16)e0[e]; pf[qb][e + 8] = (_Float16)e1[e]; }
        }

        // ---- O += P x V, V fragments from LDS shared by both q-blocks ----
        const _Float16* bv = sV[cur];
#pragma unroll
        for (int dt = 0; dt < 4; ++dt) {
            v16h vf = load_frag16(bv + (dt * 16) * VT_PAD, VT_PAD, lane);
            o[0][dt] = WMMA_F16(pf[0], vf, o[0][dt]);
            o[1][dt] = WMMA_F16(pf[1], vf, o[1][dt]);
        }

        wait_copies();
        __syncthreads();
    }

    // Normalize by row sums and store f16 at [token][head*64 + d].
#pragma unroll
    for (int qb = 0; qb < 2; ++qb) {
        float lv[8];
#pragma unroll
        for (int v = 0; v < 8; ++v) lv[v] = 1.f / __shfl(l_run[qb], h2 + v, 32);
#pragma unroll
        for (int dt = 0; dt < 4; ++dt)
#pragma unroll
            for (int v = 0; v < 8; ++v) {
                int tok = bb * SEQ + q0 + 16 * qb + h2 + v;
                int col = hd * DHEAD + dt * 16 + cl;
                AO[(size_t)tok * DIM + col] = (_Float16)(o[qb][dt][v] * lv[v]);
            }
    }
}

// ---------------------------------------------------------------------------
// 5) Output GEMM: [8192 x 1024] x [1024 x 1024] + bias -> f32 out
// ---------------------------------------------------------------------------
__global__ __launch_bounds__(32) void k_out_gemm(const _Float16* __restrict__ A,
                                                 const _Float16* __restrict__ WT,
                                                 const float* __restrict__ bias,
                                                 float* __restrict__ out) {
    const int m0 = blockIdx.x * 64;
    const int n0 = blockIdx.y * 64;
    const int lane = threadIdx.x;

    v8f acc[4][4] = {};
    for (int k0 = 0; k0 < DIM; k0 += 32) {
        v16h a[4], b[4];
#pragma unroll
        for (int i = 0; i < 4; ++i)
            a[i] = load_frag16(A + (size_t)(m0 + 16 * i) * DIM + k0, DIM, lane);
#pragma unroll
        for (int j = 0; j < 4; ++j)
            b[j] = load_frag16(WT + (size_t)(n0 + 16 * j) * DIM + k0, DIM, lane);
        if (k0 + 32 < DIM) {
            __builtin_prefetch(A + (size_t)(m0 + (lane & 15)) * DIM + k0 + 32, 0, 3);
        }
#pragma unroll
        for (int i = 0; i < 4; ++i)
#pragma unroll
            for (int j = 0; j < 4; ++j)
                acc[i][j] = WMMA_F16(a[i], b[j], acc[i][j]);
    }

    const int h2 = (lane >> 4) << 3;
    const int cl = lane & 15;
#pragma unroll
    for (int i = 0; i < 4; ++i)
#pragma unroll
        for (int j = 0; j < 4; ++j)
#pragma unroll
            for (int v = 0; v < 8; ++v) {
                int r = m0 + 16 * i + h2 + v;
                int c = n0 + 16 * j + cl;
                out[(size_t)r * DIM + c] = acc[i][j][v] + bias[c];
            }
}

// ---------------------------------------------------------------------------
// Launcher
// ---------------------------------------------------------------------------
extern "C" void kernel_launch(void* const* d_in, const int* in_sizes, int n_in,
                              void* d_out, int out_size, void* d_ws, size_t ws_size,
                              hipStream_t stream) {
    const float* x     = (const float*)d_in[0];
    const float* ln_g  = (const float*)d_in[1];
    const float* ln_b  = (const float*)d_in[2];
    const float* w_qkv = (const float*)d_in[3];
    const float* w_out = (const float*)d_in[4];
    const float* b_out = (const float*)d_in[5];
    float* out = (float*)d_out;

    // Workspace layout (bytes). Total ~72 MB. xn region is reused for attn output.
    char* ws = (char*)d_ws;
    const size_t SZ_ACT = (size_t)TOKENS * DIM * sizeof(_Float16);          // 16 MB
    _Float16* xn    = (_Float16*)(ws);                                      // + attn out later
    _Float16* Q     = (_Float16*)(ws + SZ_ACT);
    _Float16* Kd    = (_Float16*)(ws + 2 * SZ_ACT);
    _Float16* Vt    = (_Float16*)(ws + 3 * SZ_ACT);
    _Float16* wqkvT = (_Float16*)(ws + 4 * SZ_ACT);                         // 6 MB
    _Float16* woutT = (_Float16*)(ws + 4 * SZ_ACT + (size_t)DIM * QKVCOLS * sizeof(_Float16));

    // 1) Transpose + cast weights to f16 [N][K]
    k_transpose_cast<<<(DIM * QKVCOLS + 255) / 256, 256, 0, stream>>>(w_qkv, wqkvT, DIM, QKVCOLS);
    k_transpose_cast<<<(DIM * DIM + 255) / 256, 256, 0, stream>>>(w_out, woutT, DIM, DIM);

    // 2) LayerNorm -> f16
    k_layernorm<<<TOKENS, 256, 0, stream>>>(x, ln_g, ln_b, xn);

    // 3) QKV projection (WMMA)
    k_qkv_gemm<<<dim3(TOKENS / 64, QKVCOLS / 64), 32, 0, stream>>>(xn, wqkvT, Q, Kd, Vt);

    // 4) Attention (WMMA flash, LDS-shared K/V); writes f16 [tokens][1024] into xn region
    _Float16* attn_out = xn;
    k_attention<<<BATCH * HEADS * (SEQ / 128), 128, 0, stream>>>(Q, Kd, Vt, attn_out);

    // 5) Output projection + bias (WMMA)
    k_out_gemm<<<dim3(TOKENS / 64, DIM / 64), 32, 0, stream>>>(attn_out, woutT, b_out, out);
}
